// fuzzyConv_75771813036985
// MI455X (gfx1250) — matrix operand order repaired and checked by reading, not compiled, not run on hardware
//
#include <hip/hip_runtime.h>

typedef __attribute__((ext_vector_type(16))) __bf16  v16bf;
typedef __attribute__((ext_vector_type(8)))  float   v8f;
typedef __attribute__((ext_vector_type(4)))  float   v4f;
typedef __attribute__((ext_vector_type(4)))  unsigned int v4u;

union FragU {
    unsigned int u[8];
    v4u          q[2];
    v16bf        v;
};

// Round-to-nearest-even-ish bf16 pair pack: lo -> [15:0], hi -> [31:16]
__device__ __forceinline__ unsigned int pack_bf16_pair(float lo, float hi) {
    unsigned int ul = __float_as_uint(lo);
    unsigned int uh = __float_as_uint(hi);
    ul = (ul + 0x7FFFu + ((ul >> 16) & 1u)) >> 16;
    uh = (uh + 0x7FFFu + ((uh >> 16) & 1u)) & 0xFFFF0000u;
    return uh | (ul & 0xFFFFu);
}

// ---------------------------------------------------------------------------
// Prep: build B fragments (18 K-tiles, WMMA bf16 B-layout) + const term per rule.
//   feature k = (((ki*3+kj)*32)+c)*2 + w ; w=0 -> x^2 (weight s2), w=1 -> x (weight -2*s2*mu)
//   B 32x16, k order used here (contiguous, same as A): lane 0-15 -> K=0..15 (V0:K0,1 ... V7:K14,15), lane 16-31 -> K=16..31
// ---------------------------------------------------------------------------
__global__ __launch_bounds__(256) void fuzzy_prep_kernel(
    const float* __restrict__ sigma,   // (3,3,32,8)
    const float* __restrict__ mu,      // (8,32)
    unsigned int* __restrict__ wsB,    // 18*32*8 uints
    float* __restrict__ wsC)           // 8 floats
{
    const int tid = threadIdx.x;
    for (int e = tid; e < 18 * 32 * 8; e += 256) {
        int t = e >> 8;             // K-tile 0..17
        int l = (e >> 3) & 31;      // lane
        int v = e & 7;              // vgpr
        int ki = t / 6, kj = (t / 2) % 3, ch = t & 1;
        int r = l & 15;
        int cp = ((l >= 16) ? 8 : 0) + v;   // c' = K/2 within tile
        int c = ch * 16 + cp;
        float wlo = 0.0f, whi = 0.0f;
        if (r < 8) {
            float sg = sigma[((ki * 3 + kj) * 32 + c) * 8 + r];
            float s2 = sg * sg;
            float mm = mu[r * 32 + c];
            wlo = s2;               // pairs with x^2 (K even)
            whi = -2.0f * s2 * mm;  // pairs with x   (K odd)
        }
        wsB[e] = pack_bf16_pair(wlo, whi);
    }
    if (tid < 8) {
        float acc = 0.0f;
        for (int i = 0; i < 288; ++i) {     // i = (ki*3+kj)*32 + c
            int c = i & 31;
            float sg = sigma[i * 8 + tid];
            float mm = mu[tid * 32 + c];
            acc += sg * sg * mm * mm;
        }
        wsC[tid] = acc;
    }
}

// ---------------------------------------------------------------------------
// Main: implicit-GEMM fuzzy conv + fused exp/normalize/consequent.
// Block = 256 threads (8 waves). Tile = 2 output rows x 64 output cols.
// ---------------------------------------------------------------------------
#define LDSW 66   // 64 cols + 2 halo

__global__ __launch_bounds__(256) void fuzzy_main_kernel(
    const float* __restrict__ x,        // (32,256,256,32)
    const float* __restrict__ residual, // (1,)
    const float* __restrict__ cb,       // (9,16)
    const unsigned int* __restrict__ wsB,
    const float* __restrict__ wsC,
    float* __restrict__ out)            // (32,254,254,16)
{
    __shared__ unsigned int ldsX[4 * LDSW * 32];  // {bf16(x^2),bf16(x)} pairs
    __shared__ float        ldsQ[8 * 256];        // per-wave 16x16 transpose

    const int tid = threadIdx.x;
    const int b  = blockIdx.z;
    const int h0 = blockIdx.y * 2;
    const int w0 = blockIdx.x * 64;

    // ---- Stage x tile into LDS (4 rows x 66 cols x 32 ch), pre-packed bf16 pairs.
    const long xbase = (long)b * 256 * 256 * 32;
    #pragma unroll
    for (int i = 0; i < 33; ++i) {               // 33*256 == 4*66*32 exactly
        int e   = i * 256 + tid;
        int row = e / (LDSW * 32);
        int rem = e - row * (LDSW * 32);
        int col = rem >> 5;
        int c   = rem & 31;
        int gw  = w0 + col; if (gw > 255) gw = 255;   // clamp halo (edge tiles)
        float xv = x[xbase + (((long)(h0 + row) * 256 + gw) * 32 + c)];
        ldsX[e] = pack_bf16_pair(xv * xv, xv);
    }
    __syncthreads();

    const int wave   = tid >> 5;
    const int lane   = tid & 31;
    const int lanehi = lane >> 4;
    const int m      = lane & 15;        // pixel row of A (M index)
    const int wy     = wave >> 2;        // output row within tile (0..1)
    const int wxc    = (wave & 3) * 16;  // output col base within tile

    v8f acc = {0.f, 0.f, 0.f, 0.f, 0.f, 0.f, 0.f, 0.f};

    // A-layout c' offsets: lanes 0-15 -> {0..7}; lanes 16-31 -> {8..15}
    const int abase = wy * (LDSW * 32) + (wxc + m) * 32 + (lanehi ? 8 : 0);

    #pragma unroll
    for (int ki = 0; ki < 3; ++ki)
    #pragma unroll
    for (int kj = 0; kj < 3; ++kj)
    #pragma unroll
    for (int ch = 0; ch < 2; ++ch) {
        const int t = (ki * 3 + kj) * 2 + ch;

        FragU bf;
        const v4u* bp = (const v4u*)(wsB + (t * 32 + lane) * 8);
        bf.q[0] = bp[0];
        bf.q[1] = bp[1];

        FragU af;
        const int o = abase + ki * (LDSW * 32) + kj * 32 + ch * 16;
        #pragma unroll
        for (int v = 0; v < 8; ++v)
            af.u[v] = ldsX[o + v];

        acc = __builtin_amdgcn_wmma_f32_16x16x32_bf16(
            false, af.v, false, bf.v, (short)0, acc, false, false);
    }

    // ---- Transpose quad through per-wave LDS region (same-wave DS ordering).
    float* q = ldsQ + wave * 256;
    #pragma unroll
    for (int j = 0; j < 8; ++j)
        q[(j + (lanehi ? 8 : 0)) * 16 + (lane & 15)] = acc[j];
    __asm__ volatile("s_wait_dscnt 0x0" ::: "memory");

    // ---- Epilogue: 2 lanes per pixel (each does 8 of 16 outputs).
    const int   p    = lane & 15;
    const float resv = residual[0];
    float phi[8];
    float s = resv + 1e-9f;
    #pragma unroll
    for (int r = 0; r < 8; ++r) {
        phi[r] = __expf(-0.5f * (q[p * 16 + r] + wsC[r]));
        s += phi[r];
    }
    const float inv = 1.0f / s;
    const int ob = lanehi * 8;
    float o8[8];
    #pragma unroll
    for (int j = 0; j < 8; ++j) o8[j] = resv * cb[8 * 16 + ob + j];
    #pragma unroll
    for (int r = 0; r < 8; ++r) {
        const float pr = phi[r];
        #pragma unroll
        for (int j = 0; j < 8; ++j) o8[j] += pr * cb[r * 16 + ob + j];
    }

    const int h = h0 + wy;
    const int w = w0 + wxc + p;
    if (h < 254 && w < 254) {
        float* dst = out + ((((long)b * 254 + h) * 254 + w) * 16 + ob);
        v4f lo4 = { o8[0] * inv, o8[1] * inv, o8[2] * inv, o8[3] * inv };
        v4f hi4 = { o8[4] * inv, o8[5] * inv, o8[6] * inv, o8[7] * inv };
        ((v4f*)dst)[0] = lo4;
        ((v4f*)dst)[1] = hi4;
    }
}

// ---------------------------------------------------------------------------
extern "C" void kernel_launch(void* const* d_in, const int* in_sizes, int n_in,
                              void* d_out, int out_size, void* d_ws, size_t ws_size,
                              hipStream_t stream) {
    const float* x     = (const float*)d_in[0];
    const float* sigma = (const float*)d_in[1];
    const float* mu    = (const float*)d_in[2];
    const float* resid = (const float*)d_in[3];
    const float* cb    = (const float*)d_in[4];
    unsigned int* wsB  = (unsigned int*)d_ws;                 // 18*32*8 u32 = 18432 B
    float*        wsC  = (float*)((char*)d_ws + 18432);       // 8 floats
    float*        out  = (float*)d_out;

    hipLaunchKernelGGL(fuzzy_prep_kernel, dim3(1), dim3(256), 0, stream,
                       sigma, mu, wsB, wsC);

    dim3 grid(4, 127, 32);   // ceil(254/64), 254/2, batch
    hipLaunchKernelGGL(fuzzy_main_kernel, grid, dim3(256), 0, stream,
                       x, resid, cb, wsB, wsC, out);
}
